// SimpleViG_GNN_35433480192922
// MI455X (gfx1250) — matrix operand order repaired and checked by reading, not compiled, not run on hardware
//
#include <hip/hip_runtime.h>

#define N_NODES 8192
#define FEAT    192
#define HID     256
#define NCLS    1000
#define NPAD2   1024
#define KNN     9

typedef __attribute__((ext_vector_type(16))) __bf16 v16bf;
typedef __attribute__((ext_vector_type(8)))  __bf16 v8bf;
typedef __attribute__((ext_vector_type(8)))  float  v8f;

__device__ __forceinline__ unsigned short f2bf(float f) {
  unsigned u = __float_as_uint(f);
  u += 0x7FFFu + ((u >> 16) & 1u);
  return (unsigned short)(u >> 16);
}

// CDNA5 async global->LDS copy (ASYNCcnt), 16B granule.
// lds_off: wave-relative LDS byte address (low 32 bits of generic pointer).
__device__ __forceinline__ void async_load_b128(unsigned lds_off, const void* gptr) {
  asm volatile("global_load_async_to_lds_b128 %0, %1, off"
               :: "v"(lds_off), "v"((unsigned long long)(size_t)gptr)
               : "memory");
}
__device__ __forceinline__ void wait_async0() {
  asm volatile("s_wait_asynccnt 0" ::: "memory");
}
__device__ __forceinline__ unsigned lds_off_of(const void* p) {
  return (unsigned)(size_t)p;
}

// A-fragment (16x32 bf16, MxK): lane m (0-15) holds K 0-7 & 16-23 of row m,
// lane m+16 holds K 8-15 & 24-31. `base` points at the K-low 8-element run.
__device__ __forceinline__ v16bf load_a_frag(const unsigned short* base) {
  v8bf lo = *reinterpret_cast<const v8bf*>(base);
  v8bf hi = *reinterpret_cast<const v8bf*>(base + 16);
  v16bf r;
#pragma unroll
  for (int e = 0; e < 8; ++e) { r[e] = lo[e]; r[e + 8] = hi[e]; }
  return r;
}

// ---------------------------------------------------------------------------
// fp32 -> bf16 elementwise
__global__ void cvt_bf16_kernel(const float* __restrict__ x,
                                unsigned short* __restrict__ xb, int n) {
  int i = blockIdx.x * blockDim.x + threadIdx.x;
  if (i < n) xb[i] = f2bf(x[i]);
}

// per-row squared norm, one wave per row
__global__ __launch_bounds__(32) void sqnorm_kernel(const float* __restrict__ x,
                                                    float* __restrict__ sq) {
  int i = blockIdx.x;
  int l = threadIdx.x;
  float s = 0.f;
  for (int c = l; c < FEAT; c += 32) { float v = x[i * FEAT + c]; s += v * v; }
#pragma unroll
  for (int m = 16; m; m >>= 1) s += __shfl_xor(s, m, 32);
  if (l == 0) sq[i] = s;
}

// ---------------------------------------------------------------------------
// Fused kNN: WMMA Gram tiles (64 rows x 32 cols) in LDS + per-thread top-9.
// Block = 64 threads (2 waves). Thread t owns row i0+t's top-9 list.
// Tile feeds use GLOBAL_LOAD_ASYNC_TO_LDS_B128 (ASYNCcnt path).
__global__ __launch_bounds__(64) void knn_kernel(const unsigned short* __restrict__ xb,
                                                 const float* __restrict__ sq,
                                                 int* __restrict__ knn_idx) {
  __shared__ alignas(16) unsigned short Abuf[64 * FEAT];   // 24 KB row tile
  __shared__ alignas(16) unsigned short Bbuf[32 * FEAT];   // 12 KB col tile
  __shared__ float Dbuf[64 * 33];                          // stride-33: no bank conflicts
  __shared__ float sqAsh[64];
  __shared__ float sqBsh[32];

  const int tid  = threadIdx.x;       // 0..63
  const int i0   = blockIdx.x * 64;
  const int lane = tid & 31;
  const int wave = tid >> 5;          // column sub-tile (0..1)
  const int half = lane >> 4;
  const int ln   = lane & 15;

  const unsigned abase = lds_off_of(Abuf);
  const unsigned bbase = lds_off_of(Bbuf);

  // stage the 64x192 row tile once: 1536 x 16B async chunks
  for (int e = tid; e < 64 * FEAT / 8; e += 64)
    async_load_b128(abase + e * 16, xb + (size_t)i0 * FEAT + e * 8);
  if (tid < 64) sqAsh[tid] = sq[i0 + tid];

  float v[KNN];
  int   id[KNN];
#pragma unroll
  for (int s = 0; s < KNN; ++s) { v[s] = 3.0e38f; id[s] = 0; }

  for (int j0 = 0; j0 < N_NODES; j0 += 32) {
    __syncthreads();  // Dbuf consumed, Bbuf free
    // stage 32x192 col tile: 768 x 16B async chunks
    for (int e = tid; e < 32 * FEAT / 8; e += 64)
      async_load_b128(bbase + e * 16, xb + (size_t)j0 * FEAT + e * 8);
    if (tid < 32) sqBsh[tid] = sq[j0 + tid];
    wait_async0();            // our async stores to LDS are done
    __syncthreads();          // everyone's are done

    // each wave: 4 row sub-tiles x 1 col sub-tile of 16x16
    const int col = wave * 16 + ln;
    const float sqb = sqBsh[col];
#pragma unroll
    for (int mi = 0; mi < 4; ++mi) {
      v8f acc = {0.f, 0.f, 0.f, 0.f, 0.f, 0.f, 0.f, 0.f};
#pragma unroll
      for (int kc = 0; kc < FEAT; kc += 32) {
        v16bf af = load_a_frag(Abuf + (mi * 16 + ln) * FEAT + kc + half * 8);
        // B-fragment: lane n (0-15) = column n, same K order as the A-fragment
        v16bf bfr = load_a_frag(
            Bbuf + (wave * 16 + ln) * FEAT + kc + half * 8);
        acc = __builtin_amdgcn_wmma_f32_16x16x32_bf16(
            false, af, false, bfr, (short)0, acc, false, false);
      }
#pragma unroll
      for (int r = 0; r < 8; ++r) {
        int row = mi * 16 + half * 8 + r;
        float d = sqAsh[row] + sqb - 2.0f * acc[r];
        if (i0 + row == j0 + col) d = 0.0f;  // exact self-loop
        Dbuf[row * 33 + col] = d;
      }
    }
    __syncthreads();

    // thread t scans its row's 32 candidates
#pragma unroll
    for (int c = 0; c < 32; ++c) {
      float d = Dbuf[tid * 33 + c];
      if (d < v[KNN - 1]) {
        v[KNN - 1] = d; id[KNN - 1] = j0 + c;
#pragma unroll
        for (int s = KNN - 1; s > 0; --s) {
          if (v[s] < v[s - 1]) {
            float tv = v[s]; v[s] = v[s - 1]; v[s - 1] = tv;
            int   ti = id[s]; id[s] = id[s - 1]; id[s - 1] = ti;
          }
        }
      }
    }
  }

#pragma unroll
  for (int s = 0; s < KNN; ++s) knn_idx[(size_t)(i0 + tid) * KNN + s] = id[s];
}

// ---------------------------------------------------------------------------
// Pack [Wl ; Wr] transposed to bf16: wt[n][k] with zero padding for n >= cout.
__global__ void pack_w_kernel(const float* __restrict__ wl,
                              const float* __restrict__ wr,
                              unsigned short* __restrict__ wt,
                              int cin, int cout, int npad) {
  int n = blockIdx.x;  // 0..npad-1
  int K2 = 2 * cin;
  for (int k = threadIdx.x; k < K2; k += blockDim.x) {
    float val = 0.f;
    if (n < cout) val = (k < cin) ? wl[(size_t)k * cout + n]
                                  : wr[(size_t)(k - cin) * cout + n];
    wt[(size_t)n * K2 + k] = f2bf(val);
  }
}

// Gather + mean over 9 neighbors; emit bf16 [agg | self] row of length 2*cin.
__global__ void gather_kernel(const float* __restrict__ h,
                              const int* __restrict__ idx,
                              unsigned short* __restrict__ acat, int cin) {
  __shared__ int nb[KNN];
  int i = blockIdx.x;
  int c = threadIdx.x;  // blockDim == cin
  if (c < KNN) nb[c] = idx[(size_t)i * KNN + c];
  __syncthreads();
  float s = 0.f;
#pragma unroll
  for (int t = 0; t < KNN; ++t) s += h[(size_t)nb[t] * cin + c];
  s *= (1.0f / (float)KNN);
  acat[(size_t)i * (2 * cin) + c]       = f2bf(s);
  acat[(size_t)i * (2 * cin) + cin + c] = f2bf(h[(size_t)i * cin + c]);
}

// ---------------------------------------------------------------------------
// out[8192, Npad] = relu( Acat[8192,K] @ Wt^T + bias ), Wt stored [Npad][K].
// 256 threads = 8 waves in 4x2; each wave -> 32x32 output (4 WMMA accums).
// Tile staging uses GLOBAL_LOAD_ASYNC_TO_LDS_B128.
__global__ __launch_bounds__(256) void sage_gemm_kernel(
    const unsigned short* __restrict__ A, const unsigned short* __restrict__ Wt,
    const float* __restrict__ bias, float* __restrict__ out,
    int K, int Npad, int Cout) {
  __shared__ alignas(16) unsigned short Asub[128 * 32];  // 8 KB
  __shared__ alignas(16) unsigned short Bsub[64 * 32];   // 4 KB

  const int tid  = threadIdx.x;
  const int m0   = blockIdx.x * 128;
  const int n0   = blockIdx.y * 64;
  const int wid  = tid >> 5, lane = tid & 31;
  const int wm   = wid & 3,  wn   = wid >> 2;
  const int half = lane >> 4, ln  = lane & 15;

  const unsigned asub_base = lds_off_of(Asub);
  const unsigned bsub_base = lds_off_of(Bsub);

  v8f acc[2][2];
#pragma unroll
  for (int ti = 0; ti < 2; ++ti)
#pragma unroll
    for (int tj = 0; tj < 2; ++tj)
#pragma unroll
      for (int r = 0; r < 8; ++r) acc[ti][tj][r] = 0.f;

  for (int k0 = 0; k0 < K; k0 += 32) {
    __syncthreads();
    // A tile: 128 rows x 64B = 512 x 16B chunks (2 per thread)
#pragma unroll
    for (int e = tid; e < 512; e += 256) {
      int r = e >> 2, c = e & 3;
      async_load_b128(asub_base + r * 64 + c * 16,
                      A + (size_t)(m0 + r) * K + k0 + c * 8);
    }
    // W tile: 64 rows x 64B = 256 x 16B chunks (1 per thread)
    {
      int r = tid >> 2, c = tid & 3;
      async_load_b128(bsub_base + r * 64 + c * 16,
                      Wt + (size_t)(n0 + r) * K + k0 + c * 8);
    }
    wait_async0();
    __syncthreads();

    v16bf a[2], b[2];
#pragma unroll
    for (int t = 0; t < 2; ++t) {
      a[t] = load_a_frag(Asub + (wm * 32 + t * 16 + ln) * 32 + half * 8);
      b[t] = load_a_frag(
          Bsub + (wn * 32 + t * 16 + ln) * 32 + half * 8);
    }
#pragma unroll
    for (int ti = 0; ti < 2; ++ti)
#pragma unroll
      for (int tj = 0; tj < 2; ++tj)
        acc[ti][tj] = __builtin_amdgcn_wmma_f32_16x16x32_bf16(
            false, a[ti], false, b[tj], (short)0, acc[ti][tj], false, false);
  }

#pragma unroll
  for (int ti = 0; ti < 2; ++ti)
#pragma unroll
    for (int tj = 0; tj < 2; ++tj) {
      int col = n0 + wn * 32 + tj * 16 + ln;
      float bb = (col < Cout) ? bias[col] : 0.f;
#pragma unroll
      for (int r = 0; r < 8; ++r) {
        int row = m0 + wm * 32 + ti * 16 + half * 8 + r;
        float val = acc[ti][tj][r] + bb;
        out[(size_t)row * Npad + col] = val > 0.f ? val : 0.f;
      }
    }
}

// ---------------------------------------------------------------------------
// global_mean_pool: 16 segments of 512 nodes, first 1000 of 1024 cols.
__global__ void pool_kernel(const float* __restrict__ h2, float* __restrict__ out) {
  int b = blockIdx.x;
  for (int c = threadIdx.x; c < NCLS; c += blockDim.x) {
    float s = 0.f;
    const float* p = h2 + (size_t)b * 512 * NPAD2 + c;
    for (int r = 0; r < 512; ++r) s += p[(size_t)r * NPAD2];
    out[b * NCLS + c] = s * (1.0f / 512.0f);
  }
}

// ---------------------------------------------------------------------------
extern "C" void kernel_launch(void* const* d_in, const int* in_sizes, int n_in,
                              void* d_out, int out_size, void* d_ws, size_t ws_size,
                              hipStream_t stream) {
  (void)in_sizes; (void)n_in; (void)out_size; (void)ws_size;
  const float* x    = (const float*)d_in[0];
  const float* w_l0 = (const float*)d_in[1];
  const float* b_l0 = (const float*)d_in[2];
  const float* w_r0 = (const float*)d_in[3];
  const float* w_l1 = (const float*)d_in[4];
  const float* b_l1 = (const float*)d_in[5];
  const float* w_r1 = (const float*)d_in[6];
  const float* w_l2 = (const float*)d_in[7];
  const float* b_l2 = (const float*)d_in[8];
  const float* w_r2 = (const float*)d_in[9];
  float* out = (float*)d_out;

  size_t off = 0;
  auto alloc = [&](size_t bytes) -> void* {
    void* p = (void*)((char*)d_ws + off);
    off += (bytes + 255) & ~((size_t)255);
    return p;
  };
  unsigned short* xb   = (unsigned short*)alloc((size_t)N_NODES * FEAT * 2);
  float*          sq   = (float*)alloc((size_t)N_NODES * 4);
  int*            idx  = (int*)alloc((size_t)N_NODES * KNN * 4);
  unsigned short* acat = (unsigned short*)alloc((size_t)N_NODES * 2 * HID * 2);
  unsigned short* wt   = (unsigned short*)alloc((size_t)NPAD2 * 2 * HID * 2);
  float*          h0   = (float*)alloc((size_t)N_NODES * HID * 4);
  float*          h1   = (float*)alloc((size_t)N_NODES * HID * 4);
  float*          h2   = (float*)alloc((size_t)N_NODES * NPAD2 * 4);

  // graph construction
  cvt_bf16_kernel<<<(N_NODES * FEAT + 255) / 256, 256, 0, stream>>>(x, xb, N_NODES * FEAT);
  sqnorm_kernel<<<N_NODES, 32, 0, stream>>>(x, sq);
  knn_kernel<<<N_NODES / 64, 64, 0, stream>>>(xb, sq, idx);

  // layer 0: Cin=192, Cout=256
  pack_w_kernel<<<HID, 256, 0, stream>>>(w_l0, w_r0, wt, FEAT, HID, HID);
  gather_kernel<<<N_NODES, FEAT, 0, stream>>>(x, idx, acat, FEAT);
  sage_gemm_kernel<<<dim3(N_NODES / 128, HID / 64), 256, 0, stream>>>(
      acat, wt, b_l0, h0, 2 * FEAT, HID, HID);

  // layer 1: Cin=256, Cout=256
  pack_w_kernel<<<HID, 256, 0, stream>>>(w_l1, w_r1, wt, HID, HID, HID);
  gather_kernel<<<N_NODES, HID, 0, stream>>>(h0, idx, acat, HID);
  sage_gemm_kernel<<<dim3(N_NODES / 128, HID / 64), 256, 0, stream>>>(
      acat, wt, b_l1, h1, 2 * HID, HID, HID);

  // layer 2: Cin=256, Cout=1000 (padded to 1024)
  pack_w_kernel<<<NPAD2, 256, 0, stream>>>(w_l2, w_r2, wt, HID, NCLS, NPAD2);
  gather_kernel<<<N_NODES, HID, 0, stream>>>(h1, idx, acat, HID);
  sage_gemm_kernel<<<dim3(N_NODES / 128, NPAD2 / 64), 256, 0, stream>>>(
      acat, wt, b_l2, h2, 2 * HID, NPAD2, NCLS);

  pool_kernel<<<16, 256, 0, stream>>>(h2, out);
}
